// CrossAttention_59957743452639
// MI455X (gfx1250) — compile-verified, matrix operand order checked by reading, not run on hardware
//
#include <hip/hip_runtime.h>
#include <hip/hip_bf16.h>
#include <math.h>

typedef __bf16 bf16_t;
typedef __attribute__((ext_vector_type(16))) __bf16 v16bf;
typedef __attribute__((ext_vector_type(8)))  float  v8f;
typedef unsigned int u32x4 __attribute__((ext_vector_type(4)));
typedef int          i32x8 __attribute__((ext_vector_type(8)));
typedef int          i32x4 __attribute__((ext_vector_type(4)));

#if defined(__has_builtin)
#  if __has_builtin(__builtin_amdgcn_tensor_load_to_lds) && __has_builtin(__builtin_amdgcn_s_wait_tensorcnt)
#    define HAVE_TDM 1
#  endif
#endif
#ifndef HAVE_TDM
#  define HAVE_TDM 0
#endif

namespace {

constexpr int BB = 8, LL = 8, KK = 32, WW = 8;
constexpr int DIM = 256, HEADS = 8, DH = 32;
constexpr int NTOK = WW * WW;          // 64 pixels per window
constexpr int LDP  = DIM + 2;          // 258 bf16 LDS row stride (odd dword stride)
constexpr int MATS = 6;
constexpr int WMAT = DIM * DIM;        // 65536
constexpr int WELEM = NTOK * DIM;      // 16384 activations per window
constexpr float SCALE = 0.17677669529663687f;  // DH^-0.5

// LDS byte map (dynamic shared starts at offset 0)
constexpr unsigned XBYTES = NTOK * LDP * 2;        // 33024 per staging/result buffer
constexpr unsigned OFF_X0 = 0;
constexpr unsigned OFF_X1 = 1 * XBYTES;
constexpr unsigned OFF_Q  = 2 * XBYTES;
constexpr unsigned OFF_V  = 3 * XBYTES;
constexpr unsigned OFF_T  = 4 * XBYTES;
constexpr unsigned OFF_P  = 5 * XBYTES;
constexpr unsigned SMEM_BYTES = OFF_P + NTOK * HEADS * LL * 4;  // ~181.5 KB < 320 KB WGP LDS

// ---------------------------------------------------------------------------
// Prep kernel 1: slice/transpose/convert weights to bf16 W^T[j][c] in d_ws.
//   mat 0: Q_ego  mat 1: V_ego  mat 2: K_other  mat 3: V_other
//   mat 4: W_out_ego  mat 5: W_out_other
// (q_other and k_ego are dead: the others' softmax is over a size-1 axis.)
// ---------------------------------------------------------------------------
__global__ void prep_weights_kernel(const float* __restrict__ Wqkv_e,
                                    const float* __restrict__ Wqkv_o,
                                    const float* __restrict__ Wout_e,
                                    const float* __restrict__ Wout_o,
                                    bf16_t* __restrict__ Wt) {
  int idx = blockIdx.x * blockDim.x + threadIdx.x;
  if (idx >= MATS * WMAT) return;
  int mat = idx >> 16;
  int j   = (idx >> 8) & 255;
  int c   = idx & 255;
  float v;
  switch (mat) {
    case 0:  v = Wqkv_e[c * 768 + 0   + j]; break;
    case 1:  v = Wqkv_e[c * 768 + 512 + j]; break;
    case 2:  v = Wqkv_o[c * 768 + 256 + j]; break;
    case 3:  v = Wqkv_o[c * 768 + 512 + j]; break;
    case 4:  v = Wout_e[c * 256 + j];       break;
    default: v = Wout_o[c * 256 + j];       break;
  }
  Wt[idx] = (bf16_t)v;
}

// ---------------------------------------------------------------------------
// Prep kernel 2: activations fp32 [win][c][t] -> bf16 token-major [win][t][c].
// The bf16 "other" tensor (67 MB) then lives in the 192 MB L2 for both
// attention passes, and windows become dense 32 KB tiles for the TDM.
// ---------------------------------------------------------------------------
__global__ void convert_acts_kernel(const float* __restrict__ ego,
                                    const float* __restrict__ other,
                                    bf16_t* __restrict__ egoBF,
                                    bf16_t* __restrict__ otherBF) {
  constexpr long long NE = (long long)BB * KK * WELEM;       //  4,194,304
  constexpr long long NO = (long long)BB * LL * KK * WELEM;  // 33,554,432
  long long idx = (long long)blockIdx.x * blockDim.x + threadIdx.x;
  if (idx >= NO) return;
  long long w = idx >> 14;            // window
  int t = (int)((idx >> 8) & 63);
  int c = (int)(idx & 255);
  otherBF[idx] = (bf16_t)other[(w << 14) + c * 64 + t];
  if (idx < NE) egoBF[idx] = (bf16_t)ego[(idx & ~16383LL) + c * 64 + t];
}

// ---------------------------------------------------------------------------
// TDM: async 2D tile copy, 64 rows x 256 bf16 (contiguous in memory), into
// LDS with +1 dword pad per 128 dwords -> exactly the [64][258] bf16 layout.
// D# per CDNA5 ISA ch.8; tracked with TENSORcnt.
// ---------------------------------------------------------------------------
#if HAVE_TDM
__device__ inline void tdm_load_tile(unsigned lds_off, const bf16_t* gsrc) {
  unsigned long long ga = (unsigned long long)(__UINTPTR_TYPE__)gsrc;
  u32x4 g0;
  g0[0] = 1u;                                        // count=1, no gather
  g0[1] = lds_off;                                   // lds_addr
  g0[2] = (unsigned)(ga & 0xffffffffu);              // global_addr[31:0]
  g0[3] = (unsigned)((ga >> 32) & 0x1ffffffu) | (2u << 30);  // addr[56:32] | type=2
  i32x8 g1;
  g1[0] = (1 << 16) | (1 << 20) | (6 << 22);  // data_size=2B, pad_en, interval=128dw, amount=1dw
  g1[1] = (DIM & 0xffff) << 16;               // tensor_dim0 lo (256)
  g1[2] = (NTOK & 0xffff) << 16;              // tensor_dim0 hi(0) | tensor_dim1 lo (64)
  g1[3] = (DIM & 0xffff) << 16;               // tensor_dim1 hi(0) | tile_dim0 (256)
  g1[4] = NTOK;                               // tile_dim1 (64), tile_dim2=0
  g1[5] = DIM;                                // tensor_dim0_stride (256)
  g1[6] = (WELEM & 0xffff) << 16;             // dim0_stride hi(0) | tensor_dim1_stride lo
  g1[7] = WELEM >> 16;                        // tensor_dim1_stride hi (0)
  i32x4 z4 = {0, 0, 0, 0};
#if defined(__clang_major__) && (__clang_major__ >= 23)
  i32x8 z8 = {0, 0, 0, 0, 0, 0, 0, 0};
  __builtin_amdgcn_tensor_load_to_lds(g0, g1, z4, z4, z8, 0);
#else
  __builtin_amdgcn_tensor_load_to_lds(g0, g1, z4, z4, 0);
#endif
}
#endif

// ---------------------------------------------------------------------------
// WMMA fragment loaders (CDNA5 wave32 16-bit layouts).
// ---------------------------------------------------------------------------
__device__ inline v16bf load_a_frag(const bf16_t* __restrict__ sA, int t0, int kc, int lane) {
  const bf16_t* row = sA + (t0 + (lane & 15)) * LDP + kc + ((lane < 16) ? 0 : 8);
  v16bf a;
#pragma unroll
  for (int v = 0; v < 8; ++v) {
    int k = (v < 4) ? 2 * v : 16 + 2 * (v - 4);
    a[2 * v]     = row[k];
    a[2 * v + 1] = row[k + 1];
  }
  return a;
}

__device__ inline v16bf load_b_frag(const bf16_t* __restrict__ Wt, int j0, int kc, int lane) {
  const bf16_t* row = Wt + (size_t)(j0 + (lane & 15)) * DIM + kc + ((lane < 16) ? 0 : 8);
  v16bf b;
#pragma unroll
  for (int v = 0; v < 8; ++v) {
    int k = (v < 4) ? 2 * v : 16 + 2 * (v - 4);
    b[2 * v]     = row[k];
    b[2 * v + 1] = row[k + 1];
  }
  return b;
}

__device__ inline v8f gemm_tile(const bf16_t* __restrict__ sA, int t0,
                                const bf16_t* __restrict__ Wt, int j0, int lane) {
  v8f acc = {};
#pragma unroll
  for (int kc = 0; kc < DIM; kc += 32) {
    v16bf a = load_a_frag(sA, t0, kc, lane);
    v16bf b = load_b_frag(Wt, j0, kc, lane);
    acc = __builtin_amdgcn_wmma_f32_16x16x32_bf16(false, a, false, b,
                                                  (short)0, acc, false, false);
  }
  return acc;
}

__device__ inline void store_tile(bf16_t* __restrict__ dst, int t0, int j0, v8f acc, int lane) {
  const int j  = j0 + (lane & 15);
  const int tb = t0 + ((lane >> 4) << 3);
#pragma unroll
  for (int r = 0; r < 8; ++r) dst[(tb + r) * LDP + j] = (bf16_t)acc[r];
}

// ---------------------------------------------------------------------------
// Fused cross-attention: one block per (b, k) window, 8 wave32s.
// ---------------------------------------------------------------------------
__global__ __launch_bounds__(256, 1) void fused_xattn_kernel(
    const bf16_t* __restrict__ egoBF,    // [B*K][64][256] bf16
    const bf16_t* __restrict__ otherBF,  // [B*L*K][64][256] bf16
    const bf16_t* __restrict__ Wt,       // 6 x [256][256] bf16 (transposed)
    const float* __restrict__ bout_e,
    const float* __restrict__ bout_o,
    float* __restrict__ out)
{
  extern __shared__ char smem_raw[];
  bf16_t* sX0 = (bf16_t*)(smem_raw + OFF_X0);
  bf16_t* sX1 = (bf16_t*)(smem_raw + OFF_X1);
  bf16_t* sQ  = (bf16_t*)(smem_raw + OFF_Q);
  bf16_t* sV  = (bf16_t*)(smem_raw + OFF_V);
  bf16_t* sT  = (bf16_t*)(smem_raw + OFF_T);
  float*  sP  = (float*) (smem_raw + OFF_P);   // [64 tok][8 head][8 l]

  const int tid  = threadIdx.x;
  const int lane = tid & 31;
  const int wave = tid >> 5;
  const int b = blockIdx.x / KK;
  const int k = blockIdx.x % KK;

  const bf16_t* WtQe = Wt + 0 * WMAT;
  const bf16_t* WtVe = Wt + 1 * WMAT;
  const bf16_t* WtKo = Wt + 2 * WMAT;
  const bf16_t* WtVo = Wt + 3 * WMAT;
  const bf16_t* WtOe = Wt + 4 * WMAT;
  const bf16_t* WtOo = Wt + 5 * WMAT;

  const bf16_t* egoW = egoBF + (size_t)(b * KK + k) * WELEM;
  const bf16_t* othW0 = otherBF + (size_t)(b * LL * KK + k) * WELEM;  // l stride below
  const size_t lstride = (size_t)KK * WELEM;

  // this thread owns two (token, head) attention rows
  const int pr0 = tid * 2;
  const int t_a = pr0 >> 3,       m_a = pr0 & 7;
  const int t_b = (pr0 + 1) >> 3, m_b = (pr0 + 1) & 7;

  // ---- stage ego (X1) and other l=0 (X0) via TDM; project q_e / v_e ----
#if HAVE_TDM
  if (wave == 0) {
    tdm_load_tile(OFF_X1, egoW);
    tdm_load_tile(OFF_X0, othW0);
    __builtin_amdgcn_s_wait_tensorcnt(1);   // ego tile landed
  }
#else
  for (int i = tid; i < WELEM; i += 256)
    sX1[(i >> 8) * LDP + (i & 255)] = egoW[i];
#endif
  __syncthreads();
  for (int tile = wave; tile < 64; tile += 8) {
    const int t0 = (tile >> 4) * 16, j0 = (tile & 15) * 16;
    store_tile(sQ, t0, j0, gemm_tile(sX1, t0, WtQe, j0, lane), lane);
    store_tile(sV, t0, j0, gemm_tile(sX1, t0, WtVe, j0, lane), lane);
  }
  __syncthreads();

  // ---- pass 1: logits[t, m, l] = scale * q_e . k_o_l (double-buffered TDM) ----
  for (int l = 0; l < LL; ++l) {
    bf16_t* cur = (l & 1) ? sX1 : sX0;
#if HAVE_TDM
    if (wave == 0) {
      if (l < LL - 1) {
        tdm_load_tile((l & 1) ? OFF_X0 : OFF_X1, othW0 + (size_t)(l + 1) * lstride);
        __builtin_amdgcn_s_wait_tensorcnt(1);   // tile l is the oldest pending
      } else {
        __builtin_amdgcn_s_wait_tensorcnt(0);
      }
    }
#else
    {
      const bf16_t* src = othW0 + (size_t)l * lstride;
      for (int i = tid; i < WELEM; i += 256)
        cur[(i >> 8) * LDP + (i & 255)] = src[i];
    }
#endif
    __syncthreads();
    for (int tile = wave; tile < 64; tile += 8) {
      const int t0 = (tile >> 4) * 16, j0 = (tile & 15) * 16;
      store_tile(sT, t0, j0, gemm_tile(cur, t0, WtKo, j0, lane), lane);
    }
    __syncthreads();
    {
      float d0 = 0.f, d1 = 0.f;
      const bf16_t* q0 = sQ + t_a * LDP + m_a * DH;
      const bf16_t* k0 = sT + t_a * LDP + m_a * DH;
      const bf16_t* q1 = sQ + t_b * LDP + m_b * DH;
      const bf16_t* k1 = sT + t_b * LDP + m_b * DH;
#pragma unroll
      for (int c = 0; c < DH; ++c) {
        d0 += (float)q0[c] * (float)k0[c];
        d1 += (float)q1[c] * (float)k1[c];
      }
      sP[pr0 * LL + l]       = d0 * SCALE;
      sP[(pr0 + 1) * LL + l] = d1 * SCALE;
    }
    __syncthreads();
  }

  // ---- softmax over the 8 agents (each thread owns its two rows) ----
#pragma unroll
  for (int p = 0; p < 2; ++p) {
    float* lp = sP + (pr0 + p) * LL;
    float mx = lp[0];
#pragma unroll
    for (int l = 1; l < LL; ++l) mx = fmaxf(mx, lp[l]);
    float e[LL], s = 0.f;
#pragma unroll
    for (int l = 0; l < LL; ++l) { e[l] = __expf(lp[l] - mx); s += e[l]; }
    const float inv = 1.f / s;
#pragma unroll
    for (int l = 0; l < LL; ++l) lp[l] = e[l] * inv;
  }

  // ---- pass 2: o_e[t,m,:] = sum_l p[t,m,l] * v_o_l[t,m,:] (double-buffered) ----
  float oa[DH], ob[DH];
#pragma unroll
  for (int c = 0; c < DH; ++c) { oa[c] = 0.f; ob[c] = 0.f; }
#if HAVE_TDM
  if (wave == 0) tdm_load_tile(OFF_X0, othW0);
#endif
  for (int l = 0; l < LL; ++l) {
    bf16_t* cur = (l & 1) ? sX1 : sX0;
#if HAVE_TDM
    if (wave == 0) {
      if (l < LL - 1) {
        tdm_load_tile((l & 1) ? OFF_X0 : OFF_X1, othW0 + (size_t)(l + 1) * lstride);
        __builtin_amdgcn_s_wait_tensorcnt(1);
      } else {
        __builtin_amdgcn_s_wait_tensorcnt(0);
      }
    }
#else
    {
      const bf16_t* src = othW0 + (size_t)l * lstride;
      for (int i = tid; i < WELEM; i += 256)
        cur[(i >> 8) * LDP + (i & 255)] = src[i];
    }
#endif
    __syncthreads();
    for (int tile = wave; tile < 64; tile += 8) {
      const int t0 = (tile >> 4) * 16, j0 = (tile & 15) * 16;
      store_tile(sT, t0, j0, gemm_tile(cur, t0, WtVo, j0, lane), lane);
    }
    __syncthreads();
    {
      const float p0 = sP[pr0 * LL + l];
      const float p1 = sP[(pr0 + 1) * LL + l];
      const bf16_t* v0 = sT + t_a * LDP + m_a * DH;
      const bf16_t* v1 = sT + t_b * LDP + m_b * DH;
#pragma unroll
      for (int c = 0; c < DH; ++c) {
        oa[c] += p0 * (float)v0[c];
        ob[c] += p1 * (float)v1[c];
      }
    }
    __syncthreads();
  }

  // ---- park o_e in sT (bf16) for the output projection ----
#pragma unroll
  for (int c = 0; c < DH; ++c) sT[t_a * LDP + m_a * DH + c] = (bf16_t)oa[c];
#pragma unroll
  for (int c = 0; c < DH; ++c) sT[t_b * LDP + m_b * DH + c] = (bf16_t)ob[c];
  __syncthreads();

  // ---- output projections + transposed stores ----
  float* outE = out + ((size_t)b * KK + k) * (DIM * NTOK);
  float* outObase = out + (size_t)BB * KK * DIM * NTOK;
  for (int tile = wave; tile < 64; tile += 8) {
    const int t0 = (tile >> 4) * 16, j0 = (tile & 15) * 16;
    const int j  = j0 + (lane & 15);
    const int tb = t0 + ((lane >> 4) << 3);
    {
      v8f d = gemm_tile(sT, t0, WtOe, j0, lane);
      const float bias = bout_e[j];
      float4 lo4 = make_float4(d[0] + bias, d[1] + bias, d[2] + bias, d[3] + bias);
      float4 hi4 = make_float4(d[4] + bias, d[5] + bias, d[6] + bias, d[7] + bias);
      float* dst = outE + (size_t)j * NTOK + tb;
      *(float4*)(dst)     = lo4;
      *(float4*)(dst + 4) = hi4;
    }
    {
      v8f d = gemm_tile(sV, t0, WtOo, j0, lane);
      const float bias = bout_o[j];
      float4 lo4 = make_float4(d[0] + bias, d[1] + bias, d[2] + bias, d[3] + bias);
      float4 hi4 = make_float4(d[4] + bias, d[5] + bias, d[6] + bias, d[7] + bias);
#pragma unroll
      for (int l = 0; l < LL; ++l) {
        float* dst = outObase + ((((size_t)b * LL + l) * KK + k) * DIM + j) * NTOK + tb;
        *(float4*)(dst)     = lo4;
        *(float4*)(dst + 4) = hi4;
      }
    }
  }
}

}  // namespace

extern "C" void kernel_launch(void* const* d_in, const int* in_sizes, int n_in,
                              void* d_out, int out_size, void* d_ws, size_t ws_size,
                              hipStream_t stream) {
  (void)in_sizes; (void)n_in; (void)out_size; (void)ws_size;
  const float* ego    = (const float*)d_in[0];
  const float* other  = (const float*)d_in[1];
  const float* Wqkv_e = (const float*)d_in[2];
  const float* Wqkv_o = (const float*)d_in[3];
  const float* Wout_e = (const float*)d_in[4];
  const float* bout_e = (const float*)d_in[5];
  const float* Wout_o = (const float*)d_in[6];
  const float* bout_o = (const float*)d_in[7];

  // d_ws layout: 6 weight mats (768 KB) | ego bf16 (8.4 MB) | other bf16 (67 MB)
  bf16_t* Wt      = (bf16_t*)d_ws;
  bf16_t* egoBF   = Wt + (size_t)MATS * WMAT;
  bf16_t* otherBF = egoBF + (size_t)BB * KK * WELEM;

  prep_weights_kernel<<<(MATS * WMAT + 255) / 256, 256, 0, stream>>>(
      Wqkv_e, Wqkv_o, Wout_e, Wout_o, Wt);

  const long long NO = (long long)BB * LL * KK * WELEM;
  convert_acts_kernel<<<(unsigned)((NO + 255) / 256), 256, 0, stream>>>(
      ego, other, egoBF, otherBF);

  fused_xattn_kernel<<<BB * KK, 256, SMEM_BYTES, stream>>>(
      egoBF, otherBF, Wt, bout_e, bout_o, (float*)d_out);
}
